// LinearAttention_79937931313669
// MI455X (gfx1250) — matrix operand order repaired and checked by reading, not compiled, not run on hardware
//
#include <hip/hip_runtime.h>
#include <hip/hip_bf16.h>

// Problem constants (from reference): B=2, T=2048, D=1024, H=16, DH=64
#define NROWS 4096      // B*T
#define DDIM  1024
#define NHEAD 16
#define DH    64
#define CCHUNK 64
#define NCHUNK 32       // T / CCHUNK

#if defined(__HIP_DEVICE_COMPILE__) && \
    __has_builtin(__builtin_amdgcn_global_load_async_to_lds_b128) && \
    __has_builtin(__builtin_amdgcn_s_wait_asynccnt)
#define HAVE_ASYNC_LDS 1
#else
#define HAVE_ASYNC_LDS 0
#endif

typedef _Float16 v8h  __attribute__((ext_vector_type(8)));
typedef _Float16 v16h __attribute__((ext_vector_type(16)));
typedef float    v8f  __attribute__((ext_vector_type(8)));
typedef int      v4i  __attribute__((ext_vector_type(4)));

__device__ inline v8f vzero8() {
    v8f z;
#pragma unroll
    for (int i = 0; i < 8; ++i) z[i] = 0.0f;
    return z;
}

__device__ inline v16h hzero16() {
    v16h z;
#pragma unroll
    for (int i = 0; i < 16; ++i) z[i] = (_Float16)0.0f;
    return z;
}

// WMMA wrapper: D = A(16x32 f16) x B(32x16 f16) + C(16x16 f32)
__device__ inline v8f wmma_f16(v16h a, v16h b, v8f c) {
    return __builtin_amdgcn_wmma_f32_16x16x32_f16(
        /*neg_a=*/false, a, /*neg_b=*/false, b,
        /*c_mod=*/(short)0, c, /*reuse_a=*/false, /*reuse_b=*/false);
}

// A-operand loader: matrix stored row-major [m][k], row stride ld (halves).
// Contiguous k layout used here (16-bit A 16x32): lane<16 -> M=lane, halves 0..15 = K0..15
//                              lane>=16 -> M=lane-16, K16..31
__device__ inline v16h load_A16(const _Float16* base, int ld, int mBase, int k0, int lane) {
    int m = mBase + (lane & 15);
    int s = (lane >> 4) * 16;
    const _Float16* p = base + (size_t)m * ld + k0 + s;
    v8h lo = *(const v8h*)p;
    v8h hi = *(const v8h*)(p + 8);
    v16h a;
#pragma unroll
    for (int i = 0; i < 8; ++i) { a[i] = lo[i]; a[i + 8] = hi[i]; }
    return a;
}

// B-operand loader: B[k][n] = Wt[n][k] where Wt is stored row-major [n][k], row stride ld.
// Contiguous k layout used here (32x16 B): lane<16 -> N=lane, K=0..15 contiguous; lane>=16 -> N=lane-16, K=16..31.
__device__ inline v16h load_B16(const _Float16* base, int ld, int nBase, int k0, int lane) {
    int n = nBase + (lane & 15);
    int s = (lane >> 4) * 16;
    return *(const v16h*)(base + (size_t)n * ld + k0 + s);
}

#if HAVE_ASYNC_LDS
// Async global -> LDS copy of 16 bytes per lane (tracked by ASYNCcnt).
// Builtin prototype (from hipcc diagnostic): param0 = v4i in global AS.
__device__ inline void async_cp16(const _Float16* g, _Float16* l) {
    __builtin_amdgcn_global_load_async_to_lds_b128(
        (__attribute__((address_space(1))) v4i*)g,
        (__attribute__((address_space(3))) v4i*)l,
        /*offset=*/0, /*cpol=*/0);
}
#endif

// ---------------------------------------------------------------------------
// f32 -> f16 convert
// ---------------------------------------------------------------------------
__global__ void cvt_f32_to_f16(const float* __restrict__ src, _Float16* __restrict__ dst, int n) {
    int i = blockIdx.x * 256 + threadIdx.x;
    if (i < n) dst[i] = (_Float16)src[i];
}

// ---------------------------------------------------------------------------
// GEMM: out[m][j] = sum_k A[m][k] * W[j][k] + bias[j]   (A: NROWS x D, W: D x D)
// mode 0: f16 out (V) ; mode 1: f16 out with elu+1 (Q,K) ; mode 2: f32 out (final)
//
// Block tile 128x128, 8 waves as 4(M) x 2(N); each wave 32x64 (8 accumulators).
// A and B panels (128x32 f16 = 8KB each) double-buffered in LDS; 256 threads
// cooperatively stage 32B per panel per k-step (async-to-LDS if available,
// otherwise global->VGPR->ds_store).
// grid = (D/128, NROWS/128), block = 256
// ---------------------------------------------------------------------------
__global__ __launch_bounds__(256) void gemm_xWt(
    const _Float16* __restrict__ A, const _Float16* __restrict__ W,
    const float* __restrict__ bias, _Float16* __restrict__ out16,
    float* __restrict__ out32, int mode) {
    __shared__ __align__(128) _Float16 sA[2][128 * 32];
    __shared__ __align__(128) _Float16 sB[2][128 * 32];

    int tid   = threadIdx.x;
    int lane  = tid & 31;
    int wave  = tid >> 5;
    int waveM = wave & 3;         // 0..3 -> 32-row slab
    int waveN = wave >> 2;        // 0..1 -> 64-col slab
    int mBase = blockIdx.y * 128;
    int jBase = blockIdx.x * 128;

    // staging assignment: thread -> (row, 16-half segment)
    int srow = tid >> 1;          // 0..127
    int sseg = (tid & 1) * 16;    // 0 or 16
    const _Float16* gA = A + (size_t)(mBase + srow) * DDIM + sseg;
    const _Float16* gB = W + (size_t)(jBase + srow) * DDIM + sseg;
    _Float16* lA = &sA[0][0] + srow * 32 + sseg;   // buffer 0 base; +4096 for buffer 1
    _Float16* lB = &sB[0][0] + srow * 32 + sseg;

    const int KSTEPS = DDIM / 32;

#if HAVE_ASYNC_LDS
    // prologue: async-stage k-panel 0 into buffer 0
    async_cp16(gA,     lA);
    async_cp16(gA + 8, lA + 8);
    async_cp16(gB,     lB);
    async_cp16(gB + 8, lB + 8);
    __builtin_amdgcn_s_wait_asynccnt(0);
    __syncthreads();
#else
    {
        v16h ra = *(const v16h*)gA;
        v16h rb = *(const v16h*)gB;
        *(v16h*)lA = ra;
        *(v16h*)lB = rb;
    }
    __syncthreads();
#endif

    v8f acc[2][4];
#pragma unroll
    for (int mi = 0; mi < 2; ++mi)
#pragma unroll
        for (int nt = 0; nt < 4; ++nt) acc[mi][nt] = vzero8();

    for (int kk = 0; kk < KSTEPS; ++kk) {
        int cur = kk & 1;

#if HAVE_ASYNC_LDS
        if (kk + 1 < KSTEPS) {
            int k0 = (kk + 1) * 32;
            int nxt = (cur ^ 1) * 128 * 32;
            async_cp16(gA + k0,     lA + nxt);
            async_cp16(gA + k0 + 8, lA + nxt + 8);
            async_cp16(gB + k0,     lB + nxt);
            async_cp16(gB + k0 + 8, lB + nxt + 8);
            __builtin_prefetch((const void*)(gA + k0 + 32), 0, 1);
            __builtin_prefetch((const void*)(gB + k0 + 32), 0, 1);
        }
#else
        v16h ra = hzero16(), rb = hzero16();
        if (kk + 1 < KSTEPS) {
            int k0 = (kk + 1) * 32;
            ra = *(const v16h*)(gA + k0);
            rb = *(const v16h*)(gB + k0);
            __builtin_prefetch((const void*)(gA + k0 + 32), 0, 1);
            __builtin_prefetch((const void*)(gB + k0 + 32), 0, 1);
        }
#endif

        const _Float16* pA = &sA[cur][0];
        const _Float16* pB = &sB[cur][0];
        v16h a0 = load_A16(pA, 32, waveM * 32 +  0, 0, lane);
        v16h a1 = load_A16(pA, 32, waveM * 32 + 16, 0, lane);
#pragma unroll
        for (int nt = 0; nt < 4; ++nt) {
            v16h b = load_B16(pB, 32, waveN * 64 + nt * 16, 0, lane);
            acc[0][nt] = wmma_f16(a0, b, acc[0][nt]);
            acc[1][nt] = wmma_f16(a1, b, acc[1][nt]);
        }

#if HAVE_ASYNC_LDS
        if (kk + 1 < KSTEPS) __builtin_amdgcn_s_wait_asynccnt(0);
        __syncthreads();               // publish next buffer; release current
#else
        __syncthreads();               // everyone done reading buffers
        if (kk + 1 < KSTEPS) {
            int nxt = (cur ^ 1) * 128 * 32;
            *(v16h*)(lA + nxt) = ra;
            *(v16h*)(lB + nxt) = rb;
        }
        __syncthreads();               // publish next buffer
#endif
    }

    // epilogue: bias + optional elu+1, scatter stores
#pragma unroll
    for (int mi = 0; mi < 2; ++mi) {
#pragma unroll
        for (int nt = 0; nt < 4; ++nt) {
            int j = jBase + waveN * 64 + nt * 16 + (lane & 15);
            float bj = bias[j];
#pragma unroll
            for (int r = 0; r < 8; ++r) {
                int m = mBase + waveM * 32 + mi * 16 + r + ((lane >> 4) << 3);
                float v = acc[mi][nt][r] + bj;
                if (mode == 1) v = (v > 0.0f) ? (v + 1.0f) : __expf(v);  // elu(x)+1
                if (mode == 2) out32[(size_t)m * DDIM + j] = v;
                else           out16[(size_t)m * DDIM + j] = (_Float16)v;
            }
        }
    }
}

// ---------------------------------------------------------------------------
// Chunked causal linear attention. One workgroup per (b,h); 4 waves walk
// NCHUNK chunks of 64 timesteps, keeping S (64x64, f32) and z (64, f32) in LDS.
// ---------------------------------------------------------------------------
__global__ __launch_bounds__(128) void linattn_chunk(
    const _Float16* __restrict__ Qf, const _Float16* __restrict__ Kf,
    const _Float16* __restrict__ Vf, _Float16* __restrict__ Of) {
    __shared__ __align__(128) _Float16 sVt[CCHUNK * CCHUNK];   // V transposed: [e][t]
    __shared__ __align__(128) _Float16 sKt[CCHUNK * CCHUNK];   // K transposed: [d][t]
    __shared__ __align__(128) _Float16 sSc[CCHUNK * CCHUNK];   // masked scores [i][j]
    __shared__ __align__(128) _Float16 sS16[CCHUNK * CCHUNK];  // f16 copy of state [e][d]
    __shared__ float sS32[CCHUNK * CCHUNK];                    // f32 state St[e][d]=S[d][e]
    __shared__ float sZ[CCHUNK];
    __shared__ float sDen[CCHUNK];

    int tid  = threadIdx.x;
    int lane = tid & 31;
    int wv   = tid >> 5;          // 0..3
    int b    = blockIdx.x / NHEAD;
    int h    = blockIdx.x % NHEAD;
    size_t hOff = (size_t)h * DH;

    for (int i = tid; i < CCHUNK * CCHUNK; i += 128) sS32[i] = 0.0f;
    if (tid < CCHUNK) sZ[tid] = 0.0f;
    __syncthreads();

    for (int c = 0; c < NCHUNK; ++c) {
        size_t n0 = (size_t)b * 2048 + (size_t)c * CCHUNK;
        const _Float16* Qc = Qf + n0 * DDIM + hOff;
        const _Float16* Kc = Kf + n0 * DDIM + hOff;
        const _Float16* Vc = Vf + n0 * DDIM + hOff;
        _Float16*       Oc = Of + n0 * DDIM + hOff;

        // Stage transposed V, K and f16 state copy
        for (int i = tid; i < CCHUNK * CCHUNK; i += 128) {
            int t = i >> 6, e = i & 63;
            sVt[e * CCHUNK + t] = Vc[(size_t)t * DDIM + e];
            sKt[e * CCHUNK + t] = Kc[(size_t)t * DDIM + e];
            sS16[i] = (_Float16)sS32[i];
        }
        // den part 1: q_i . z_prev
        if (tid < CCHUNK) {
            float s = 0.0f;
            for (int d = 0; d < DH; ++d)
                s += (float)Qc[(size_t)tid * DDIM + d] * sZ[d];
            sDen[tid] = s;
        }
        __syncthreads();

        // Scores = Qf @ Kf^T, causal-masked (j <= i), written to LDS as f16
        {
            int mt = wv;
#pragma unroll
            for (int nt = 0; nt < 4; ++nt) {
                v8f acc = vzero8();
                for (int k0 = 0; k0 < DH; k0 += 32) {
                    v16h a = load_A16(Qc, DDIM, mt * 16, k0, lane);
                    v16h bb = load_B16(Kc, DDIM, nt * 16, k0, lane);
                    acc = wmma_f16(a, bb, acc);
                }
                int j = nt * 16 + (lane & 15);
#pragma unroll
                for (int r = 0; r < 8; ++r) {
                    int i2 = mt * 16 + r + ((lane >> 4) << 3);
                    float v = (j <= i2) ? acc[r] : 0.0f;
                    sSc[i2 * CCHUNK + j] = (_Float16)v;
                }
            }
        }
        __syncthreads();

        // den part 2 (causal row-sum) + z update (column sums of K = rows of Kt)
        if (tid < CCHUNK) {
            float rs = 0.0f;
            for (int j = 0; j < CCHUNK; ++j) rs += (float)sSc[tid * CCHUNK + j];
            float dsum = sDen[tid] + rs;
            sDen[tid] = (dsum > 1e-6f) ? dsum : 1e-6f;
            float za = 0.0f;
            for (int t = 0; t < CCHUNK; ++t) za += (float)sKt[tid * CCHUNK + t];
            sZ[tid] += za;
        }
        __syncthreads();

        // O = (Sc @ V + Qf @ S_prev) / den
        {
            int mt = wv;
#pragma unroll
            for (int nt = 0; nt < 4; ++nt) {
                v8f acc = vzero8();
                for (int k0 = 0; k0 < DH; k0 += 32) {
                    v16h a1 = load_A16(sSc, CCHUNK, mt * 16, k0, lane);   // intra
                    v16h b1 = load_B16(sVt, CCHUNK, nt * 16, k0, lane);
                    acc = wmma_f16(a1, b1, acc);
                    v16h a2 = load_A16(Qc, DDIM, mt * 16, k0, lane);      // inter
                    v16h b2 = load_B16(sS16, CCHUNK, nt * 16, k0, lane);
                    acc = wmma_f16(a2, b2, acc);
                }
                int e = nt * 16 + (lane & 15);
#pragma unroll
                for (int r = 0; r < 8; ++r) {
                    int i2 = mt * 16 + r + ((lane >> 4) << 3);
                    float o = acc[r] / sDen[i2];
                    Oc[(size_t)i2 * DDIM + e] = (_Float16)o;
                }
            }
        }

        // State update: St[e][d] += (V^T @ K)[e][d]
        {
            int et = wv;
#pragma unroll
            for (int dt = 0; dt < 4; ++dt) {
                v8f acc = vzero8();
                for (int t0 = 0; t0 < CCHUNK; t0 += 32) {
                    v16h a = load_A16(sVt, CCHUNK, et * 16, t0, lane);
                    v16h bb = load_B16(sKt, CCHUNK, dt * 16, t0, lane);
                    acc = wmma_f16(a, bb, acc);
                }
                int d = dt * 16 + (lane & 15);
#pragma unroll
                for (int r = 0; r < 8; ++r) {
                    int e = et * 16 + r + ((lane >> 4) << 3);
                    sS32[e * CCHUNK + d] += acc[r];
                }
            }
        }
        __syncthreads();
    }
}

// ---------------------------------------------------------------------------
extern "C" void kernel_launch(void* const* d_in, const int* in_sizes, int n_in,
                              void* d_out, int out_size, void* d_ws, size_t ws_size,
                              hipStream_t stream) {
    const float* x  = (const float*)d_in[0];
    const float* Wq = (const float*)d_in[1];
    const float* bq = (const float*)d_in[2];
    const float* Wk = (const float*)d_in[3];
    const float* bk = (const float*)d_in[4];
    const float* Wv = (const float*)d_in[5];
    const float* bv = (const float*)d_in[6];
    const float* Wo = (const float*)d_in[7];
    const float* bo = (const float*)d_in[8];
    float* out = (float*)d_out;

    char* ws = (char*)d_ws;
    const size_t ND  = (size_t)NROWS * DDIM;   // 4M elems
    const size_t DD  = (size_t)DDIM * DDIM;    // 1M elems
    _Float16* xh  = (_Float16*)(ws);
    _Float16* Wqh = (_Float16*)(ws + ND * 2);
    _Float16* Wkh = (_Float16*)(ws + ND * 2 + DD * 2);
    _Float16* Wvh = (_Float16*)(ws + ND * 2 + DD * 4);
    _Float16* Woh = (_Float16*)(ws + ND * 2 + DD * 6);
    _Float16* Qf  = (_Float16*)(ws + ND * 2 + DD * 8);
    _Float16* Kf  = (_Float16*)(ws + ND * 4 + DD * 8);
    _Float16* Vf  = (_Float16*)(ws + ND * 6 + DD * 8);
    _Float16* Of  = (_Float16*)(ws + ND * 8 + DD * 8);

    // 1) convert inputs to f16
    cvt_f32_to_f16<<<(int)(ND / 256), 256, 0, stream>>>(x,  xh,  (int)ND);
    cvt_f32_to_f16<<<(int)(DD / 256), 256, 0, stream>>>(Wq, Wqh, (int)DD);
    cvt_f32_to_f16<<<(int)(DD / 256), 256, 0, stream>>>(Wk, Wkh, (int)DD);
    cvt_f32_to_f16<<<(int)(DD / 256), 256, 0, stream>>>(Wv, Wvh, (int)DD);
    cvt_f32_to_f16<<<(int)(DD / 256), 256, 0, stream>>>(Wo, Woh, (int)DD);

    // 2) projections (Q,K with fused elu+1 feature map; V plain)
    dim3 gg(DDIM / 128, NROWS / 128);
    gemm_xWt<<<gg, 256, 0, stream>>>(xh, Wqh, bq, Qf, nullptr, 1);
    gemm_xWt<<<gg, 256, 0, stream>>>(xh, Wkh, bk, Kf, nullptr, 1);
    gemm_xWt<<<gg, 256, 0, stream>>>(xh, Wvh, bv, Vf, nullptr, 0);

    // 3) chunked causal linear attention: one workgroup per (b,h)
    linattn_chunk<<<dim3(2 * NHEAD), 128, 0, stream>>>(Qf, Kf, Vf, Of);

    // 4) output projection -> f32 d_out
    gemm_xWt<<<gg, 256, 0, stream>>>(Of, Woh, bo, nullptr, out, 2);
}
